// NCELoss_3925600109314
// MI455X (gfx1250) — matrix operand order repaired and checked by reading, not compiled, not run on hardware
//
#include <hip/hip_runtime.h>
#include <hip/hip_bf16.h>

// NCE loss, memory-bound: 436 MB of gathered fp32 weight rows -> ~19us floor
// at 23.3 TB/s HBM. Logits via v_wmma_f32_16x16x32_f16 diagonal-extraction
// GEMMs (16-sample tile x 26 slots); WMMA compute ~1us, off the critical path.
// Gathered rows are single-consumer (one wave, one slot), so they are staged
// in registers (compiler pipelines global_load_b128 ~13 deep) — not LDS.
// global_prefetch of the NEXT slot's rows warms the 192MB L2 a full GEMM
// ahead of use (rows average ~2.1 touches across the dispatch).

typedef __attribute__((ext_vector_type(16))) _Float16 v16h;
typedef __attribute__((ext_vector_type(8)))  _Float16 v8h;
typedef __attribute__((ext_vector_type(8)))  float    v8f;

#define EDIM   1024
#define KNOISE 25
#define NSLOT  26          // 1 target + 25 noise
#define TILE   16          // samples per workgroup
#define XS     1032        // padded f16 row stride in LDS (2064 B: 16B-aligned, bank-friendly)
#define LGS    28          // padded logits row stride

__global__ __launch_bounds__(256)
void nce_wmma_kernel(const float* __restrict__ x,
                     const int*   __restrict__ target,
                     const int*   __restrict__ noise_idx,
                     const float* __restrict__ weight,
                     const float* __restrict__ bias,
                     const float* __restrict__ noise,
                     float* __restrict__ out,
                     int n_total)
{
    __shared__ __align__(16) _Float16 xh[TILE * XS];   // x tile as f16 (~32 KB)
    __shared__ float lg[TILE * LGS];                   // logits[sample][slot]
    __shared__ float red[TILE];

    const int tid = threadIdx.x;
    const int n0  = blockIdx.x * TILE;

    // ---- Stage A: load x tile (16 x 1024 f32), convert to f16 into LDS ----
    const float4* xv = (const float4*)(x + (size_t)n0 * EDIM);
    for (int i = tid; i < TILE * EDIM / 4; i += 256) {
        float4 v = xv[i];
        int e   = i * 4;
        int row = e >> 10;          // /1024
        int col = e & (EDIM - 1);
        _Float16* dst = &xh[row * XS + col];
        dst[0] = (_Float16)v.x; dst[1] = (_Float16)v.y;
        dst[2] = (_Float16)v.z; dst[3] = (_Float16)v.w;
    }
    __syncthreads();

    const int lane = tid & 31;
    const int wave = tid >> 5;      // 8 waves per block
    const int j    = lane & 15;     // sample within tile (A row / B column)
    const int hi   = lane >> 4;     // half-wave selector

    // ---- Stage B: per-slot 16x16x1024 GEMM, keep the diagonal ----
    for (int k = wave; k < NSLOT; k += 8) {           // uniform per wave
        const int n = n0 + j;
        const int r = (k == 0) ? target[n] : noise_idx[n * KNOISE + (k - 1)];
        const float* wp = weight + (size_t)r * EDIM + hi * 16;
        const float bj = bias[r];

        // Next slot this wave will process: prefetch its gathered row into L2
        // while the current 32-chunk GEMM runs (prefetch uses no counters).
        const int k2 = (k + 8 < NSLOT) ? (k + 8) : k;
        const int r2 = (k2 == 0) ? target[n] : noise_idx[n * KNOISE + (k2 - 1)];
        const float* wp2 = weight + (size_t)r2 * EDIM + hi * 16;

        v8f c = {};
        #pragma unroll 4
        for (int ch = 0; ch < EDIM / 32; ++ch) {
            // A fragment (16x32 f16): lane holds row j, K = hi*16 + 0..15
            const _Float16* ap = &xh[j * XS + ch * 32 + hi * 16];
            v8h alo = *(const v8h*)(ap);
            v8h ahi = *(const v8h*)(ap + 8);
            v16h a = __builtin_shufflevector(alo, ahi,
                         0,1,2,3,4,5,6,7,8,9,10,11,12,13,14,15);

            // B fragment (32x16 f16): lane holds column j, K = hi*16 + 0..15,
            // converted on the fly from the gathered fp32 weight row.
            const float4* bp = (const float4*)(wp + ch * 32);
            float4 q0 = bp[0], q1 = bp[1], q2 = bp[2], q3 = bp[3];
            v16h b;
            b[0]=(_Float16)q0.x;  b[1]=(_Float16)q0.y;  b[2]=(_Float16)q0.z;  b[3]=(_Float16)q0.w;
            b[4]=(_Float16)q1.x;  b[5]=(_Float16)q1.y;  b[6]=(_Float16)q1.z;  b[7]=(_Float16)q1.w;
            b[8]=(_Float16)q2.x;  b[9]=(_Float16)q2.y;  b[10]=(_Float16)q2.z; b[11]=(_Float16)q2.w;
            b[12]=(_Float16)q3.x; b[13]=(_Float16)q3.y; b[14]=(_Float16)q3.z; b[15]=(_Float16)q3.w;

            // Warm L2 for the next slot's row segment (global_prefetch_b8).
            __builtin_prefetch(wp2 + ch * 32, 0, 2);

            c = __builtin_amdgcn_wmma_f32_16x16x32_f16(
                    /*neg_a=*/false, a, /*neg_b=*/false, b,
                    /*c_mod=*/(short)0, c, /*reuse_a=*/false, /*reuse_b=*/false);
        }

        // Diagonal: m=0..7 -> lane m, c[m]; m=8..15 -> lane m+16, c[m-8].
        float d = 0.0f;
        #pragma unroll
        for (int m = 0; m < 8; ++m) {
            float cm = c[m];
            if (lane == m || lane == m + 24) d = cm;
        }
        if (lane < 8)        lg[lane * LGS + k]        = d + bj;  // sample = lane (== j)
        else if (lane >= 24) lg[(lane - 16) * LGS + k] = d + bj;  // sample = lane-16 (== j)
    }
    __syncthreads();

    // ---- Stage C: per-sample loss, block reduce, atomic accumulate ----
    if (tid < TILE) {
        const int n = n0 + tid;
        const float K = (float)KNOISE;
        float pnt = noise[target[n]];
        float p0  = expf(lg[tid * LGS + 0] - 9.0f);
        float acc = logf(p0 / (p0 + K * pnt));
        for (int k = 0; k < KNOISE; ++k) {
            float pn = noise[noise_idx[n * KNOISE + k]];
            float pk = expf(lg[tid * LGS + 1 + k] - 9.0f);
            acc += logf(K * pn / (pk + K * pn));
        }
        red[tid] = acc;
    }
    __syncthreads();
    if (tid == 0) {
        float s = 0.0f;
        #pragma unroll
        for (int i = 0; i < TILE; ++i) s += red[i];
        atomicAdd(out, -s / (float)n_total);
    }
}

extern "C" void kernel_launch(void* const* d_in, const int* in_sizes, int n_in,
                              void* d_out, int out_size, void* d_ws, size_t ws_size,
                              hipStream_t stream)
{
    const float* x         = (const float*)d_in[0];   // (N, E) f32
    const int*   target    = (const int*)  d_in[1];   // (N,)   i32
    const int*   noise_idx = (const int*)  d_in[2];   // (N, K) i32
    const float* weight    = (const float*)d_in[3];   // (V, E) f32
    const float* bias      = (const float*)d_in[4];   // (V,)   f32
    const float* noise     = (const float*)d_in[5];   // (V,)   f32
    float* out = (float*)d_out;

    const int N = in_sizes[1];                        // 4096

    hipMemsetAsync(out, 0, sizeof(float), stream);    // graph-capture safe
    nce_wmma_kernel<<<N / TILE, 256, 0, stream>>>(
        x, target, noise_idx, weight, bias, noise, out, N);
}
